// DeepFM_vcmi_42047729828461
// MI455X (gfx1250) — matrix operand order repaired and checked by reading, not compiled, not run on hardware
//
#include <hip/hip_runtime.h>
#include <hip/hip_bf16.h>
#include <math.h>

// ---------------------------------------------------------------------------
// DeepFM forward for MI455X (gfx1250, wave32, WMMA)
// ---------------------------------------------------------------------------

#define BATCH   65536
#define NSLOT   14
#define SUMEMB  46
#define SLOTE   16
#define DEEPF   (NSLOT * SLOTE)   // 224
#define HDIM    1024
#define OUTK    7
#define EPSBN   1e-5f
#define MBLOCKS (BATCH / 128)     // 512 M-tiles -> stats partials

typedef _Float16 half_t;
typedef __attribute__((ext_vector_type(16))) _Float16 v16h;
typedef __attribute__((ext_vector_type(8)))  _Float16 v8h;
typedef __attribute__((ext_vector_type(4)))  _Float16 v4h;
typedef __attribute__((ext_vector_type(8)))  float    v8f;

// ---------------------------------------------------------------------------
// Weight transpose + f32->f16 convert:  Wt[n*K + k] = (f16) W[k*N + n]
// ---------------------------------------------------------------------------
__global__ __launch_bounds__(256) void conv_transpose_f16(
    const float* __restrict__ W, half_t* __restrict__ Wt, int K, int N) {
  int i = blockIdx.x * 256 + threadIdx.x;
  if (i < K * N) {
    int n = i / K;
    int k = i - n * K;
    Wt[i] = (half_t)W[(size_t)k * N + n];
  }
}

// ---------------------------------------------------------------------------
// Embedding gather * value, shared slot linear + relu -> deep f16 [B,224]
// one thread per (batch, slot); W_slot staged in LDS
// ---------------------------------------------------------------------------
__global__ __launch_bounds__(256) void embed_slot_kernel(
    const int* __restrict__ Xi, const float* __restrict__ Xv,
    const float* __restrict__ e0, const float* __restrict__ e1,
    const float* __restrict__ e2, const float* __restrict__ e3,
    const float* __restrict__ e4, const float* __restrict__ e5,
    const float* __restrict__ e6, const float* __restrict__ e7,
    const float* __restrict__ e8, const float* __restrict__ e9,
    const float* __restrict__ Wslot, const float* __restrict__ bslot,
    half_t* __restrict__ deep) {
  __shared__ float sW[SUMEMB * SLOTE];
  __shared__ float sB[SLOTE];
  for (int i = threadIdx.x; i < SUMEMB * SLOTE; i += 256) sW[i] = Wslot[i];
  if (threadIdx.x < SLOTE) sB[threadIdx.x] = bslot[threadIdx.x];
  __syncthreads();

  int t = blockIdx.x * 256 + threadIdx.x;         // < B*14 exactly
  int b = t / NSLOT;
  int s = t - b * NSLOT;
  const int*   xi = Xi + (size_t)t * 10;
  const float* xv = Xv + (size_t)t * 10;

  float e[SUMEMB];
  {
    const float* p = e0 + xi[0] * 10;
    float v = xv[0];
    #pragma unroll
    for (int i = 0; i < 10; ++i) e[i] = p[i] * v;
  }
#define GATHER4(J, TAB, OFF)                        \
  {                                                 \
    const float* p = TAB + xi[J] * 4;               \
    float v = xv[J];                                \
    _Pragma("unroll")                               \
    for (int i = 0; i < 4; ++i) e[OFF + i] = p[i] * v; \
  }
  GATHER4(1, e1, 10) GATHER4(2, e2, 14) GATHER4(3, e3, 18)
  GATHER4(4, e4, 22) GATHER4(5, e5, 26) GATHER4(6, e6, 30)
  GATHER4(7, e7, 34) GATHER4(8, e8, 38) GATHER4(9, e9, 42)
#undef GATHER4

  half_t* out = deep + (size_t)b * DEEPF + s * SLOTE;
  #pragma unroll
  for (int h = 0; h < SLOTE; ++h) {
    float acc = sB[h];
    #pragma unroll
    for (int f = 0; f < SUMEMB; ++f) acc = fmaf(e[f], sW[f * SLOTE + h], acc);
    out[h] = (half_t)fmaxf(acc, 0.0f);
  }
}

// ---------------------------------------------------------------------------
// WMMA f16 GEMM with fused BN partial stats.
//   C[B,1024] = A[B,K] x Bt^T       (Bt stored as [N][K] f16)
// block = 256 threads = 8 waves; block tile 128(M) x 128(N); wave tile 16x128.
// Epilogue: per-block column sums / sumsqs -> psum/psq [MBLOCKS][1024]
// (fixed-order reductions: shfl across half16 lane pair, LDS tree over waves)
// Fragment layouts per CDNA5 ISA 7.12.2 (wave32).
// ---------------------------------------------------------------------------
__global__ __launch_bounds__(256) void gemm_wmma_f16_stats(
    const half_t* __restrict__ A, const half_t* __restrict__ Bt,
    float* __restrict__ C, float* __restrict__ psum, float* __restrict__ psq,
    int K) {
  const int lane   = threadIdx.x & 31;
  const int wave   = threadIdx.x >> 5;
  const int half16 = lane >> 4;      // 0: lanes 0-15, 1: lanes 16-31
  const int l15    = lane & 15;
  const int tileM  = blockIdx.x * 128 + wave * 16;
  const int tileN  = blockIdx.y * 128;

  // A: lane holds row (tileM+l15); K-runs [k0+8*half16 ..+7], [+16 ..+23]
  const half_t* aptr = A + (size_t)(tileM + l15) * K + 8 * half16;
  // B: lane holds col (tileN+16j+l15); K-runs [k0+8*half16 ..+7], [+16 ..+23]
  const half_t* bptr = Bt + (size_t)(tileN + l15) * K + 8 * half16;
  const size_t bstep = (size_t)16 * K;   // next 16-column group

  v8f acc[8];
  #pragma unroll
  for (int j = 0; j < 8; ++j) acc[j] = (v8f){};

  for (int k0 = 0; k0 < K; k0 += 32) {
    __builtin_prefetch(aptr + k0 + 32, 0, 1);   // global_prefetch_b8
    __builtin_prefetch(bptr + k0 + 32, 0, 1);

    v8h alo = *reinterpret_cast<const v8h*>(aptr + k0);
    v8h ahi = *reinterpret_cast<const v8h*>(aptr + k0 + 16);
    v16h a;
    #pragma unroll
    for (int i = 0; i < 8; ++i) { a[i] = alo[i]; a[i + 8] = ahi[i]; }

    #pragma unroll
    for (int j = 0; j < 8; ++j) {
      v8h blo = *reinterpret_cast<const v8h*>(bptr + (size_t)j * bstep + k0);
      v8h bhi = *reinterpret_cast<const v8h*>(bptr + (size_t)j * bstep + k0 + 16);
      v16h b;
      #pragma unroll
      for (int i = 0; i < 8; ++i) { b[i] = blo[i]; b[i + 8] = bhi[i]; }
      acc[j] = __builtin_amdgcn_wmma_f32_16x16x32_f16(
          false, a, false, b, (short)0, acc[j], false, false);
    }
  }

  // ---- store C; C/D layout: lane l15 = N, vgpr v -> M = tileM+v+8*half16 ----
  float* crow = C + (size_t)(tileM + 8 * half16) * HDIM + tileN + l15;
  #pragma unroll
  for (int j = 0; j < 8; ++j) {
    #pragma unroll
    for (int v = 0; v < 8; ++v) crow[(size_t)v * HDIM + 16 * j] = acc[j][v];
  }

  // ---- fused BN partial stats over this block's 128 rows ----
  __shared__ float sred[8][128];
  __shared__ float qred[8][128];
  #pragma unroll
  for (int j = 0; j < 8; ++j) {
    float sj = 0.f, qj = 0.f;
    #pragma unroll
    for (int v = 0; v < 8; ++v) { float x = acc[j][v]; sj += x; qj += x * x; }
    // combine lane L with L^16 (same column, other 8 rows) -> 16-row sums
    sj += __shfl_xor(sj, 16, 32);
    qj += __shfl_xor(qj, 16, 32);
    if (half16 == 0) {
      sred[wave][j * 16 + l15] = sj;
      qred[wave][j * 16 + l15] = qj;
    }
  }
  __syncthreads();
  if (threadIdx.x < 128) {
    float S = 0.f, Q = 0.f;
    #pragma unroll
    for (int w = 0; w < 8; ++w) { S += sred[w][threadIdx.x]; Q += qred[w][threadIdx.x]; }
    size_t o = (size_t)blockIdx.x * HDIM + tileN + threadIdx.x;
    psum[o] = S;
    psq[o]  = Q;
  }
}

// Fold MBLOCKS partials -> per-column affine BN coeffs: y = z*a + bsh
__global__ __launch_bounds__(256) void colstats_finalize(
    const float* __restrict__ psum, const float* __restrict__ psq,
    const float* __restrict__ g, const float* __restrict__ be,
    float* __restrict__ a, float* __restrict__ bsh) {
  int c = blockIdx.x * 256 + threadIdx.x;   // 1024 threads total
  float s = 0.f, q = 0.f;
  for (int p = 0; p < MBLOCKS; ++p) {
    s += psum[(size_t)p * HDIM + c];
    q += psq[(size_t)p * HDIM + c];
  }
  float m   = s * (1.0f / BATCH);
  float var = q * (1.0f / BATCH) - m * m;
  float ga  = g[c] * rsqrtf(var + EPSBN);
  a[c]   = ga;
  bsh[c] = be[c] - m * ga;
}

// Fused BN(affine) + ReLU + f32->f16:  h = relu(z*a + bsh)
__global__ __launch_bounds__(256) void bn_relu_f16(
    const float* __restrict__ z, const float* __restrict__ a,
    const float* __restrict__ bsh, half_t* __restrict__ h) {
  size_t i = (size_t)blockIdx.x * 256 + threadIdx.x;   // in float4 units
  int c4 = (int)(i & 255);
  float4 zv = reinterpret_cast<const float4*>(z)[i];
  float4 av = reinterpret_cast<const float4*>(a)[c4];
  float4 bv = reinterpret_cast<const float4*>(bsh)[c4];
  v4h o;
  o[0] = (half_t)fmaxf(fmaf(zv.x, av.x, bv.x), 0.f);
  o[1] = (half_t)fmaxf(fmaf(zv.y, av.y, bv.y), 0.f);
  o[2] = (half_t)fmaxf(fmaf(zv.z, av.z, bv.z), 0.f);
  o[3] = (half_t)fmaxf(fmaf(zv.w, av.w, bv.w), 0.f);
  reinterpret_cast<v4h*>(h)[i] = o;
}

// ---------------------------------------------------------------------------
// Heads: one wave per row, 8 dot-products of length 1024.
// head[b][0] = h.Ww ; head[b][1..7] = h.Wk[:,k]
// ---------------------------------------------------------------------------
__global__ __launch_bounds__(256) void heads_kernel(
    const half_t* __restrict__ h, const float* __restrict__ Wk,
    const float* __restrict__ Ww, float* __restrict__ head) {
  int row  = (blockIdx.x * 256 + threadIdx.x) >> 5;
  int lane = threadIdx.x & 31;
  const half_t* hr = h + (size_t)row * HDIM;
  float acc[8] = {0.f, 0.f, 0.f, 0.f, 0.f, 0.f, 0.f, 0.f};
  for (int i = lane; i < HDIM; i += 32) {
    float hv = (float)hr[i];
    acc[0] = fmaf(hv, Ww[i], acc[0]);
    const float* wk = Wk + (size_t)i * OUTK;
    #pragma unroll
    for (int k = 0; k < OUTK; ++k) acc[k + 1] = fmaf(hv, wk[k], acc[k + 1]);
  }
  #pragma unroll
  for (int k = 0; k < 8; ++k)
    for (int off = 16; off > 0; off >>= 1)
      acc[k] += __shfl_xor(acc[k], off, 32);
  if (lane == 0) {
    float* o = head + (size_t)row * 8;
    #pragma unroll
    for (int k = 0; k < 8; ++k) o[k] = acc[k];
  }
}

// head column partial stats: 128 blocks x 256 -> per-wave partials [1024][16]
__global__ __launch_bounds__(256) void headstats_partial(
    const float* __restrict__ head, float* __restrict__ hpart) {
  int gid = blockIdx.x * 256 + threadIdx.x;   // 32768 threads
  float s[8] = {0}, q[8] = {0};
  for (int r = gid; r < BATCH; r += 32768) {
    const float* p = head + (size_t)r * 8;
    #pragma unroll
    for (int k = 0; k < 8; ++k) { float v = p[k]; s[k] += v; q[k] += v * v; }
  }
  #pragma unroll
  for (int k = 0; k < 8; ++k)
    for (int off = 16; off > 0; off >>= 1) {
      s[k] += __shfl_xor(s[k], off, 32);
      q[k] += __shfl_xor(q[k], off, 32);
    }
  int lane = threadIdx.x & 31;
  if (lane == 0) {
    float* o = hpart + (size_t)(gid >> 5) * 16;
    #pragma unroll
    for (int k = 0; k < 8; ++k) { o[k] = s[k]; o[8 + k] = q[k]; }
  }
}

__global__ void headstats_finalize(
    const float* __restrict__ hpart,
    const float* __restrict__ gk, const float* __restrict__ bek,
    const float* __restrict__ gw, const float* __restrict__ bew,
    float* __restrict__ ha, float* __restrict__ hb) {
  int c = threadIdx.x;
  if (c >= 8) return;
  float s = 0.f, q = 0.f;
  for (int p = 0; p < 1024; ++p) {
    s += hpart[(size_t)p * 16 + c];
    q += hpart[(size_t)p * 16 + 8 + c];
  }
  float m   = s * (1.0f / BATCH);
  float var = q * (1.0f / BATCH) - m * m;
  float rstd = rsqrtf(var + EPSBN);
  float g  = (c == 0) ? gw[0]  : gk[c - 1];
  float be = (c == 0) ? bew[0] : bek[c - 1];
  ha[c] = g * rstd;
  hb[c] = be - m * g * rstd;
}

// final: win = sigmoid(BN(head0)) -> out[0..B); kill = BN(head1..7) -> out[B..)
__global__ __launch_bounds__(256) void final_out_kernel(
    const float* __restrict__ head, const float* __restrict__ ha,
    const float* __restrict__ hb, float* __restrict__ out) {
  int b = blockIdx.x * 256 + threadIdx.x;
  const float* p = head + (size_t)b * 8;
  float w = fmaf(p[0], ha[0], hb[0]);
  out[b] = 1.0f / (1.0f + expf(-w));
  float* ko = out + BATCH + (size_t)b * OUTK;
  #pragma unroll
  for (int k = 0; k < OUTK; ++k) ko[k] = fmaf(p[k + 1], ha[k + 1], hb[k + 1]);
}

// ---------------------------------------------------------------------------
extern "C" void kernel_launch(void* const* d_in, const int* in_sizes, int n_in,
                              void* d_out, int out_size, void* d_ws,
                              size_t ws_size, hipStream_t stream) {
  (void)in_sizes; (void)n_in; (void)out_size; (void)ws_size;

  const int*   Xi    = (const int*)d_in[0];
  const float* Xv    = (const float*)d_in[1];
  const float* emb[10];
  for (int j = 0; j < 10; ++j) emb[j] = (const float*)d_in[2 + j];
  const float* Wslot = (const float*)d_in[12];
  const float* bslot = (const float*)d_in[13];
  const float* W1    = (const float*)d_in[14];   // b1 (15) absorbed by BN
  const float* g1    = (const float*)d_in[16];
  const float* be1   = (const float*)d_in[17];
  const float* W2    = (const float*)d_in[18];   // b2 (19) absorbed
  const float* g2    = (const float*)d_in[20];
  const float* be2   = (const float*)d_in[21];
  const float* Wk    = (const float*)d_in[22];   // bk (23) absorbed
  const float* gk    = (const float*)d_in[24];
  const float* bek   = (const float*)d_in[25];
  const float* Ww    = (const float*)d_in[26];   // bw (27) absorbed
  const float* gw    = (const float*)d_in[28];
  const float* bew   = (const float*)d_in[29];
  float* out = (float*)d_out;

  // workspace carve-out (256B aligned)
  char* base = (char*)d_ws;
  size_t off = 0;
  auto carve = [&](size_t bytes) {
    char* p = base + off;
    off += (bytes + 255) & ~(size_t)255;
    return p;
  };
  half_t* W1t   = (half_t*)carve((size_t)HDIM * DEEPF * 2);     // [1024][224]
  half_t* W2t   = (half_t*)carve((size_t)HDIM * HDIM * 2);      // [1024][1024]
  half_t* deep  = (half_t*)carve((size_t)BATCH * DEEPF * 2);
  float*  z     = (float*)carve((size_t)BATCH * HDIM * 4);      // reused L1/L2
  half_t* h     = (half_t*)carve((size_t)BATCH * HDIM * 2);     // reused h1/h2
  float*  head  = (float*)carve((size_t)BATCH * 8 * 4);
  float*  psum  = (float*)carve((size_t)MBLOCKS * HDIM * 4);
  float*  psq   = (float*)carve((size_t)MBLOCKS * HDIM * 4);
  float*  acoef = (float*)carve((size_t)HDIM * 4);
  float*  bcoef = (float*)carve((size_t)HDIM * 4);
  float*  hpart = (float*)carve((size_t)1024 * 16 * 4);
  float*  ha    = (float*)carve(64);
  float*  hb    = (float*)carve(64);

  // 1) weight transpose+convert (tiny, amortized)
  conv_transpose_f16<<<(DEEPF * HDIM + 255) / 256, 256, 0, stream>>>(
      W1, W1t, DEEPF, HDIM);
  conv_transpose_f16<<<(HDIM * HDIM + 255) / 256, 256, 0, stream>>>(
      W2, W2t, HDIM, HDIM);

  // 2) embedding + slot linear -> deep f16 [B,224]
  embed_slot_kernel<<<(BATCH * NSLOT) / 256, 256, 0, stream>>>(
      Xi, Xv, emb[0], emb[1], emb[2], emb[3], emb[4], emb[5], emb[6], emb[7],
      emb[8], emb[9], Wslot, bslot, deep);

  // 3) layer 1: WMMA GEMM (fused stats) -> z, BN coeffs, BN+relu -> h (f16)
  gemm_wmma_f16_stats<<<dim3(BATCH / 128, HDIM / 128), 256, 0, stream>>>(
      deep, W1t, z, psum, psq, DEEPF);
  colstats_finalize<<<HDIM / 256, 256, 0, stream>>>(psum, psq, g1, be1,
                                                    acoef, bcoef);
  bn_relu_f16<<<(BATCH * HDIM / 4) / 256, 256, 0, stream>>>(z, acoef, bcoef, h);

  // 4) layer 2
  gemm_wmma_f16_stats<<<dim3(BATCH / 128, HDIM / 128), 256, 0, stream>>>(
      h, W2t, z, psum, psq, HDIM);
  colstats_finalize<<<HDIM / 256, 256, 0, stream>>>(psum, psq, g2, be2,
                                                    acoef, bcoef);
  bn_relu_f16<<<(BATCH * HDIM / 4) / 256, 256, 0, stream>>>(z, acoef, bcoef, h);

  // 5) heads + head BN + sigmoid -> d_out (win [B], then kill [B,7])
  heads_kernel<<<(BATCH * 32) / 256, 256, 0, stream>>>(h, Wk, Ww, head);
  headstats_partial<<<128, 256, 0, stream>>>(head, hpart);
  headstats_finalize<<<1, 32, 0, stream>>>(hpart, gk, bek, gw, bew, ha, hb);
  final_out_kernel<<<BATCH / 256, 256, 0, stream>>>(head, ha, hb, out);
}
